// RewardAgentEncoder_26164940767431
// MI455X (gfx1250) — matrix operand order repaired and checked by reading, not compiled, not run on hardware
//
#include <hip/hip_runtime.h>

typedef _Float16 h16;
typedef __attribute__((ext_vector_type(16))) _Float16 v16h;
typedef __attribute__((ext_vector_type(8)))  float    v8f;

#define DEV __device__ __forceinline__

constexpr int   HDIM    = 128;
constexpr int   FIN     = 129;   // 2*64+1 fourier features
constexpr int   FPAD    = 160;   // padded to multiple of 32 for WMMA K
constexpr float PI_F    = 3.14159265358979323846f;
constexpr float TWOPI_F = 6.28318530717958647692f;

enum { EPI_F32 = 0, EPI_F16 = 1, EPI_RELU_F16 = 2, EPI_SIG_F32 = 3,
       EPI_ACC_F32 = 4, EPI_LNRELU_F16 = 5 };

// ---------------------------------------------------------------------------
// Core WMMA GEMM:  C[M x N] = epi( A[M x K] (f16,row-major) * B[K x N] + bias )
// B is passed pre-transposed as Bt[N x K] (f16).  One wave -> 16 rows x 128
// cols; 4 waves/block share the 32x128 B-tile staged through LDS.
// M (grid.x*64) must cover padded rows; buffers are allocated padded.
// ---------------------------------------------------------------------------
__global__ void __launch_bounds__(128) k_gemm(
    const h16* __restrict__ A, int lda,
    const h16* __restrict__ Bt, int K, int Ncols,
    const float* __restrict__ bias,
    void* __restrict__ Cout, float* __restrict__ Cacc,
    const float* __restrict__ ln_s, const float* __restrict__ ln_b, int epi)
{
  __shared__ h16 Bs[128 * 32];
  const int t    = threadIdx.x;
  const int wv   = t >> 5;
  const int lane = t & 31;
  const int hl   = lane >> 4;   // which 16-lane half
  const int cl   = lane & 15;
  const int rowBase = (blockIdx.x * 4 + wv) * 16;
  const int colBase = blockIdx.y * 128;

  v8f acc[8] = {};

  for (int kc = 0; kc < K; kc += 32) {
    __syncthreads();
    { // cooperative stage of B^T tile: thread t loads 32 contiguous K-values of col t
      const h16* s = Bt + (size_t)(colBase + t) * K + kc;
      h16* d = Bs + t * 32;
#pragma unroll
      for (int i = 0; i < 32; ++i) d[i] = s[i];
    }
    __syncthreads();

    // A fragment (16x32 f16): lanes 0-15 rows, K 0-7 & 16-23; lanes 16-31 K 8-15 & 24-31
    v16h af;
    {
      const h16* p = A + (size_t)(rowBase + cl) * lda + kc + hl * 8;
#pragma unroll
      for (int i = 0; i < 8; ++i) af[i] = p[i];
#pragma unroll
      for (int i = 0; i < 8; ++i) af[8 + i] = p[16 + i];
    }
#pragma unroll
    for (int nt = 0; nt < 8; ++nt) {
      // B fragment (32x16): lane holds one column; lanes 0-15 K 0-7 & 16-23, lanes 16-31 K 8-15 & 24-31
      v16h bf;
      const h16* p = Bs + (nt * 16 + cl) * 32 + hl * 8;
#pragma unroll
      for (int i = 0; i < 8; ++i) { bf[i] = p[i]; bf[8 + i] = p[16 + i]; }
      acc[nt] = __builtin_amdgcn_wmma_f32_16x16x32_f16(
          false, af, false, bf, (short)0, acc[nt], false, false);
    }
  }

  if (bias) {
#pragma unroll
    for (int nt = 0; nt < 8; ++nt) {
      float bv = bias[colBase + nt * 16 + cl];
#pragma unroll
      for (int v = 0; v < 8; ++v) acc[nt][v] += bv;
    }
  }

  if (epi == EPI_LNRELU_F16) {
    // fused row LayerNorm: row r's 128 values live in one 16-lane half across 8 tiles
#pragma unroll
    for (int v = 0; v < 8; ++v) {
      float s = 0.f, ss = 0.f;
#pragma unroll
      for (int nt = 0; nt < 8; ++nt) { float x = acc[nt][v]; s += x; ss += x * x; }
#pragma unroll
      for (int m = 1; m < 16; m <<= 1) { s += __shfl_xor(s, m, 32); ss += __shfl_xor(ss, m, 32); }
      float mean = s * (1.f / 128.f);
      float rstd = rsqrtf(ss * (1.f / 128.f) - mean * mean + 1e-5f);
#pragma unroll
      for (int nt = 0; nt < 8; ++nt) {
        int col = colBase + nt * 16 + cl;
        float x = (acc[nt][v] - mean) * rstd * ln_s[col] + ln_b[col];
        acc[nt][v] = fmaxf(x, 0.f);
      }
    }
  } else if (epi == EPI_RELU_F16) {
#pragma unroll
    for (int nt = 0; nt < 8; ++nt)
#pragma unroll
      for (int v = 0; v < 8; ++v) acc[nt][v] = fmaxf(acc[nt][v], 0.f);
  } else if (epi == EPI_SIG_F32) {
#pragma unroll
    for (int nt = 0; nt < 8; ++nt)
#pragma unroll
      for (int v = 0; v < 8; ++v) acc[nt][v] = 1.f / (1.f + expf(-acc[nt][v]));
  }

#pragma unroll
  for (int nt = 0; nt < 8; ++nt) {
#pragma unroll
    for (int v = 0; v < 8; ++v) {
      size_t idx = (size_t)(rowBase + v + hl * 8) * Ncols + colBase + nt * 16 + cl;
      float x = acc[nt][v];
      if (epi == EPI_F32 || epi == EPI_SIG_F32)      ((float*)Cout)[idx] = x;
      else if (epi == EPI_ACC_F32)                   Cacc[idx] += x;
      else                                           ((h16*)Cout)[idx] = (h16)x;
    }
  }
}

// ---------------------------------------------------------------------------
// Row LayerNorm helpers (1 wave = 1 row of 128)
// ---------------------------------------------------------------------------
DEV float warp_sum32(float x) {
#pragma unroll
  for (int m = 1; m < 32; m <<= 1) x += __shfl_xor(x, m, 32);
  return x;
}

__global__ void k_ln(const float* __restrict__ X, const float* __restrict__ s,
                     const float* __restrict__ b, h16* __restrict__ Y, int M)
{
  int row = blockIdx.x * 8 + (threadIdx.x >> 5);
  if (row >= M) return;
  int lane = threadIdx.x & 31;
  int c = lane * 4;
  const float4 v = *(const float4*)(X + (size_t)row * 128 + c);
  float mean = warp_sum32(v.x + v.y + v.z + v.w) * (1.f / 128.f);
  float sq   = warp_sum32(v.x * v.x + v.y * v.y + v.z * v.z + v.w * v.w) * (1.f / 128.f);
  float rstd = rsqrtf(sq - mean * mean + 1e-5f);
  h16* y = Y + (size_t)row * 128 + c;
  y[0] = (h16)((v.x - mean) * rstd * s[c + 0] + b[c + 0]);
  y[1] = (h16)((v.y - mean) * rstd * s[c + 1] + b[c + 1]);
  y[2] = (h16)((v.z - mean) * rstd * s[c + 2] + b[c + 2]);
  y[3] = (h16)((v.w - mean) * rstd * s[c + 3] + b[c + 3]);
}

__global__ void k_add_ln(const float* __restrict__ base, const float* __restrict__ X,
                         const float* __restrict__ s, const float* __restrict__ b,
                         float* __restrict__ Y, int M)
{
  int row = blockIdx.x * 8 + (threadIdx.x >> 5);
  if (row >= M) return;
  int lane = threadIdx.x & 31;
  int c = lane * 4;
  const float4 v  = *(const float4*)(X    + (size_t)row * 128 + c);
  const float4 bs = *(const float4*)(base + (size_t)row * 128 + c);
  float mean = warp_sum32(v.x + v.y + v.z + v.w) * (1.f / 128.f);
  float sq   = warp_sum32(v.x * v.x + v.y * v.y + v.z * v.z + v.w * v.w) * (1.f / 128.f);
  float rstd = rsqrtf(sq - mean * mean + 1e-5f);
  float* y = Y + (size_t)row * 128 + c;
  y[0] = bs.x + (v.x - mean) * rstd * s[c + 0] + b[c + 0];
  y[1] = bs.y + (v.y - mean) * rstd * s[c + 1] + b[c + 1];
  y[2] = bs.z + (v.z - mean) * rstd * s[c + 2] + b[c + 2];
  y[3] = bs.w + (v.w - mean) * rstd * s[c + 3] + b[c + 3];
}

// ACC (+ optional category embedding) -> LN -> relu -> f16
__global__ void k_postln(const float* __restrict__ ACC, const int* __restrict__ type,
                         const float* __restrict__ temb,
                         const float* __restrict__ s, const float* __restrict__ b,
                         h16* __restrict__ Y, int M)
{
  int row = blockIdx.x * 8 + (threadIdx.x >> 5);
  if (row >= M) return;
  int lane = threadIdx.x & 31;
  int c = lane * 4;
  float4 v = *(const float4*)(ACC + (size_t)row * 128 + c);
  if (type) {
    int ty = type[row];
    const float4 e = *(const float4*)(temb + (size_t)ty * 128 + c);
    v.x += e.x; v.y += e.y; v.z += e.z; v.w += e.w;
  }
  float mean = warp_sum32(v.x + v.y + v.z + v.w) * (1.f / 128.f);
  float sq   = warp_sum32(v.x * v.x + v.y * v.y + v.z * v.z + v.w * v.w) * (1.f / 128.f);
  float rstd = rsqrtf(sq - mean * mean + 1e-5f);
  h16* y = Y + (size_t)row * 128 + c;
  y[0] = (h16)fmaxf((v.x - mean) * rstd * s[c + 0] + b[c + 0], 0.f);
  y[1] = (h16)fmaxf((v.y - mean) * rstd * s[c + 1] + b[c + 1], 0.f);
  y[2] = (h16)fmaxf((v.z - mean) * rstd * s[c + 2] + b[c + 2], 0.f);
  y[3] = (h16)fmaxf((v.w - mean) * rstd * s[c + 3] + b[c + 3], 0.f);
}

// ---------------------------------------------------------------------------
// Small elementwise / feature kernels
// ---------------------------------------------------------------------------
__global__ void k_fill(float* __restrict__ p, long n, float v)
{
  long t = (long)blockIdx.x * blockDim.x + threadIdx.x;
  if (t < n) p[t] = v;
}

// transpose + f32->f16 (+ K zero-pad): dst[N x Kpad] from src[K x N]
__global__ void k_wt(const float* __restrict__ src, h16* __restrict__ dst,
                     int K, int N, int Kpad)
{
  int t = blockIdx.x * blockDim.x + threadIdx.x;
  if (t >= N * Kpad) return;
  int n = t / Kpad, k = t - n * Kpad;
  dst[t] = (k < K) ? (h16)src[(size_t)k * N + n] : (h16)0.f;
}

DEV float wrapf(float a)
{
  float m = fmodf(a + PI_F, TWOPI_F);
  if (m < 0.f) m += TWOPI_F;
  return m - PI_F;
}

__global__ void k_agent_feat(const float* __restrict__ ad, float* __restrict__ xc, int n)
{
  int t = blockIdx.x * blockDim.x + threadIdx.x;
  if (t >= n) return;
  const float* a = ad + (size_t)t * 5;
  float hx, hy; sincosf(a[2], &hy, &hx);
  float vx = a[3], vy = a[4];
  xc[(size_t)t * 2 + 0] = sqrtf(vx * vx + vy * vy);
  xc[(size_t)t * 2 + 1] = atan2f(hx * vy - hy * vx, hx * vx + hy * vy);
}

__global__ void k_edge_r1(const float* __restrict__ pos_pl, const float* __restrict__ or_pl,
                          const float* __restrict__ ad, const int* __restrict__ es,
                          const int* __restrict__ ed, float* __restrict__ r, int E)
{
  int t = blockIdx.x * blockDim.x + threadIdx.x;
  if (t >= E) return;
  int s = es[t], d = ed[t];
  const float* a = ad + (size_t)d * 5;
  float px = pos_pl[(size_t)s * 2 + 0] - a[0];
  float py = pos_pl[(size_t)s * 2 + 1] - a[1];
  float ha = a[2];
  float hx, hy; sincosf(ha, &hy, &hx);
  r[(size_t)t * 3 + 0] = sqrtf(px * px + py * py);
  r[(size_t)t * 3 + 1] = atan2f(hx * py - hy * px, hx * px + hy * py);
  r[(size_t)t * 3 + 2] = wrapf(or_pl[s] - ha);
}

__global__ void k_edge_r2(const float* __restrict__ ad, const int* __restrict__ es,
                          const int* __restrict__ ed, float* __restrict__ r, int E)
{
  int t = blockIdx.x * blockDim.x + threadIdx.x;
  if (t >= E) return;
  int s = es[t], d = ed[t];
  const float* as = ad + (size_t)s * 5;
  const float* adst = ad + (size_t)d * 5;
  float px = as[0] - adst[0];
  float py = as[1] - adst[1];
  float ha = adst[2];
  float hx, hy; sincosf(ha, &hy, &hx);
  r[(size_t)t * 3 + 0] = sqrtf(px * px + py * py);
  r[(size_t)t * 3 + 1] = atan2f(hx * py - hy * px, hx * px + hy * py);
  r[(size_t)t * 3 + 2] = wrapf(as[2] - ha);
}

// fourier features for one input dim: F[row, 0..160) = [cos, sin, x, pad0]
__global__ void k_fourier_f(const float* __restrict__ cont, int dim, int d,
                            const float* __restrict__ freqs, h16* __restrict__ F, int M)
{
  int t = blockIdx.x * blockDim.x + threadIdx.x;
  if (t >= M * 64) return;
  int row = t >> 6, j = t & 63;
  float x  = cont[(size_t)row * dim + d];
  float ph = x * freqs[d * 64 + j] * TWOPI_F;
  float sn, cs; sincosf(ph, &sn, &cs);
  size_t base = (size_t)row * FPAD;
  F[base + j]      = (h16)cs;
  F[base + 64 + j] = (h16)sn;
  if (j == 0)  F[base + 128] = (h16)x;
  if (j < 31)  F[base + 129 + j] = (h16)0.f;
}

// ---------------------------------------------------------------------------
// Edge-softmax attention kernels
// ---------------------------------------------------------------------------
DEV void atomicMaxF(float* addr, float val)
{
  int old = __float_as_int(*addr);
  while (__int_as_float(old) < val) {
    int prev = atomicCAS((int*)addr, old, __float_as_int(val));
    if (prev == old) break;
    old = prev;
  }
}

__global__ void k_sim(const h16* __restrict__ Q, const h16* __restrict__ Kb,
                      const h16* __restrict__ KR, const int* __restrict__ es,
                      const int* __restrict__ ed, float* __restrict__ sim,
                      float* __restrict__ mbuf, int E)
{
  int t = blockIdx.x * blockDim.x + threadIdx.x;
  if (t >= E * 8) return;
  int e = t >> 3, h = t & 7;
  int si = es[e], d = ed[e];
  const h16* q  = Q  + (size_t)d  * 128 + h * 16;
  const h16* k  = Kb + (size_t)si * 128 + h * 16;
  const h16* kr = KR + (size_t)e  * 128 + h * 16;
  float acc = 0.f;
#pragma unroll
  for (int i = 0; i < 16; ++i) acc += (float)q[i] * ((float)k[i] + (float)kr[i]);
  acc *= 0.25f;  // 1/sqrt(16)
  sim[t] = acc;
  atomicMaxF(mbuf + (size_t)d * 8 + h, acc);
}

__global__ void k_exp(float* __restrict__ sim, const int* __restrict__ ed,
                      const float* __restrict__ mbuf, float* __restrict__ den, int E)
{
  int t = blockIdx.x * blockDim.x + threadIdx.x;
  if (t >= E * 8) return;
  int e = t >> 3, h = t & 7;
  int d = ed[e];
  float v = expf(sim[t] - mbuf[(size_t)d * 8 + h]);
  sim[t] = v;
  atomicAdd(den + (size_t)d * 8 + h, v);
}

__global__ void k_msg(const float* __restrict__ ex, const float* __restrict__ den,
                      const h16* __restrict__ V, const h16* __restrict__ VR,
                      const int* __restrict__ es, const int* __restrict__ ed,
                      float* __restrict__ msg, int E)
{
  long t = (long)blockIdx.x * blockDim.x + threadIdx.x;
  if (t >= (long)E * 128) return;
  int e = (int)(t >> 7), ii = (int)(t & 127), h = ii >> 4;
  int si = es[e], d = ed[e];
  float a  = ex[(size_t)e * 8 + h] / den[(size_t)d * 8 + h];
  float vv = (float)V[(size_t)si * 128 + ii] + (float)VR[(size_t)e * 128 + ii];
  atomicAdd(msg + (size_t)d * 128 + ii, a * vv);
}

__global__ void k_concat(const float* __restrict__ msg, const h16* __restrict__ XD,
                         h16* __restrict__ CAT, int M)
{
  long t = (long)blockIdx.x * blockDim.x + threadIdx.x;
  if (t >= (long)M * 128) return;
  int row = (int)(t >> 7), ii = (int)(t & 127);
  CAT[(size_t)row * 256 + ii]       = (h16)msg[t];
  CAT[(size_t)row * 256 + 128 + ii] = XD[t];
}

__global__ void k_agg(const float* __restrict__ msg, const float* __restrict__ g,
                      const float* __restrict__ s, h16* __restrict__ AGG, int M)
{
  long t = (long)blockIdx.x * blockDim.x + threadIdx.x;
  if (t >= (long)M * 128) return;
  float m = msg[t];
  AGG[t] = (h16)(m + g[t] * (s[t] - m));
}

__global__ void k_gather(const float* __restrict__ X, const int* __restrict__ ego,
                         float* __restrict__ out, int n)
{
  int t = blockIdx.x * blockDim.x + threadIdx.x;
  if (t >= n * 128) return;
  int i = t >> 7, c = t & 127;
  out[t] = X[(size_t)ego[i] * 128 + c];
}

// ---------------------------------------------------------------------------
// Host side: parameter pytree unpack (jax tree_flatten = sorted dict keys)
// ---------------------------------------------------------------------------
struct FourW {
  const float *b1, *b2, *freqs, *ln1_b, *ln1_s, *out_b, *out_w, *oln_b, *oln_s, *w1, *w2;
  h16 *W1t, *W2t, *Outt;
  int dim;
};

struct AttnW {
  const float *ap_b, *ap_s, *ff1_b, *ff1_w, *ff2_b, *ff2_w,
              *ffpo_b, *ffpo_s, *ffpr_b, *ffpr_s,
              *lnd_b, *lnd_s, *lnr_b, *lnr_s, *lns_b, *lns_s,
              *tg_b, *tg_w, *tk_w, *tkr_w, *tout_b, *tout_w,
              *tq_b, *tq_w, *ts_b, *ts_w, *tv_w, *tvr_w;
  h16 *Tq, *Tk, *Tv, *Tkr, *Tvr, *Ts, *Tg, *Tout, *Tff1, *Tff2;
};

static void parse_four(void* const* in, int& i, FourW& f)
{
  f.b1    = (const float*)in[i++];  f.b2    = (const float*)in[i++];
  f.freqs = (const float*)in[i++];
  f.ln1_b = (const float*)in[i++];  f.ln1_s = (const float*)in[i++];
  f.out_b = (const float*)in[i++];  f.out_w = (const float*)in[i++];
  f.oln_b = (const float*)in[i++];  f.oln_s = (const float*)in[i++];
  f.w1    = (const float*)in[i++];  f.w2    = (const float*)in[i++];
}

static void parse_attn(void* const* in, int& i, AttnW& w, bool bip)
{
  w.ap_b  = (const float*)in[i++];  w.ap_s  = (const float*)in[i++];
  w.ff1_b = (const float*)in[i++];  w.ff1_w = (const float*)in[i++];
  w.ff2_b = (const float*)in[i++];  w.ff2_w = (const float*)in[i++];
  w.ffpo_b= (const float*)in[i++];  w.ffpo_s= (const float*)in[i++];
  w.ffpr_b= (const float*)in[i++];  w.ffpr_s= (const float*)in[i++];
  if (bip) { w.lnd_b = (const float*)in[i++]; w.lnd_s = (const float*)in[i++]; }
  else     { w.lnd_b = nullptr;               w.lnd_s = nullptr; }
  w.lnr_b = (const float*)in[i++];  w.lnr_s = (const float*)in[i++];
  w.lns_b = (const float*)in[i++];  w.lns_s = (const float*)in[i++];
  w.tg_b  = (const float*)in[i++];  w.tg_w  = (const float*)in[i++];
  w.tk_w  = (const float*)in[i++];
  w.tkr_w = (const float*)in[i++];
  w.tout_b= (const float*)in[i++];  w.tout_w= (const float*)in[i++];
  w.tq_b  = (const float*)in[i++];  w.tq_w  = (const float*)in[i++];
  w.ts_b  = (const float*)in[i++];  w.ts_w  = (const float*)in[i++];
  w.tv_w  = (const float*)in[i++];
  w.tvr_w = (const float*)in[i++];
}

extern "C" void kernel_launch(void* const* d_in, const int* in_sizes, int n_in,
                              void* d_out, int out_size, void* d_ws, size_t ws_size,
                              hipStream_t stream)
{
  (void)n_in; (void)out_size; (void)ws_size;

  const float* agents    = (const float*)d_in[0];
  const int*   atype     = (const int*)  d_in[1];
  const float* pos_pl    = (const float*)d_in[2];
  const float* orient_pl = (const float*)d_in[3];
  const float* x_pl      = (const float*)d_in[4];
  const int*   es1       = (const int*)  d_in[5];
  const int*   ed1       = (const int*)  d_in[6];
  const int*   es2       = (const int*)  d_in[7];
  const int*   ed2       = (const int*)  d_in[8];
  const int*   ego       = (const int*)  d_in[9];

  const int NA  = in_sizes[1];
  const int NPL = in_sizes[3];
  const int E1  = in_sizes[5];
  const int E2  = in_sizes[7];
  const int NEG = in_sizes[9];
  const int EMX = (E1 > E2) ? E1 : E2;

  int pi = 10;
  AttnW a2a[2], pl2a[2];
  FourW fr2, fr1, fx;
  parse_attn(d_in, pi, a2a[0],  false);
  parse_attn(d_in, pi, a2a[1],  false);
  parse_attn(d_in, pi, pl2a[0], true);
  parse_attn(d_in, pi, pl2a[1], true);
  parse_four(d_in, pi, fr2); fr2.dim = 3;
  parse_four(d_in, pi, fr1); fr1.dim = 3;
  const float* type_emb = (const float*)d_in[pi++];
  parse_four(d_in, pi, fx);  fx.dim = 2;

  auto pad64 = [](int m) { return (m + 63) & ~63; };
  auto cdiv  = [](long a, long b) { return (int)((a + b - 1) / b); };
  const int NAp = pad64(NA), EP = pad64(EMX);

  size_t off = 0;
  auto alloc = [&](size_t bytes) -> void* {
    void* p = (char*)d_ws + off;
    off = (off + bytes + 255) & ~(size_t)255;
    return p;
  };

  // ---- weight conversion (f32 -> f16, transposed, padded) ----
  auto convw = [&](const float* w, int K, int N, int Kpad) -> h16* {
    h16* d = (h16*)alloc((size_t)N * Kpad * sizeof(h16));
    int n = N * Kpad;
    k_wt<<<cdiv(n, 256), 256, 0, stream>>>(w, d, K, N, Kpad);
    return d;
  };
  auto conv_four = [&](FourW& f) {
    f.W1t = (h16*)alloc((size_t)f.dim * HDIM * FPAD * sizeof(h16));
    for (int d = 0; d < f.dim; ++d)
      k_wt<<<cdiv(HDIM * FPAD, 256), 256, 0, stream>>>(
          f.w1 + (size_t)d * FIN * HDIM, f.W1t + (size_t)d * HDIM * FPAD, FIN, HDIM, FPAD);
    f.W2t = (h16*)alloc((size_t)f.dim * HDIM * HDIM * sizeof(h16));
    for (int d = 0; d < f.dim; ++d)
      k_wt<<<cdiv(HDIM * HDIM, 256), 256, 0, stream>>>(
          f.w2 + (size_t)d * HDIM * HDIM, f.W2t + (size_t)d * HDIM * HDIM, HDIM, HDIM, HDIM);
    f.Outt = convw(f.out_w, HDIM, HDIM, HDIM);
  };
  auto conv_attn = [&](AttnW& w) {
    w.Tq   = convw(w.tq_w,   128, 128, 128);
    w.Tk   = convw(w.tk_w,   128, 128, 128);
    w.Tv   = convw(w.tv_w,   128, 128, 128);
    w.Tkr  = convw(w.tkr_w,  128, 128, 128);
    w.Tvr  = convw(w.tvr_w,  128, 128, 128);
    w.Ts   = convw(w.ts_w,   128, 128, 128);
    w.Tg   = convw(w.tg_w,   256, 128, 256);
    w.Tout = convw(w.tout_w, 128, 128, 128);
    w.Tff1 = convw(w.ff1_w,  128, 512, 128);
    w.Tff2 = convw(w.ff2_w,  512, 128, 512);
  };
  conv_four(fx); conv_four(fr1); conv_four(fr2);
  for (int l = 0; l < 2; ++l) { conv_attn(pl2a[l]); conv_attn(a2a[l]); }

  // ---- workspace buffers (all GEMM buffers padded to 64-row multiples) ----
  float* xcont = (float*)alloc((size_t)NA  * 2 * 4);
  float* r1c   = (float*)alloc((size_t)E1  * 3 * 4);
  float* r2c   = (float*)alloc((size_t)E2  * 3 * 4);
  h16*   Fbuf  = (h16*)  alloc((size_t)EP  * FPAD * 2);
  h16*   Hbuf  = (h16*)  alloc((size_t)EP  * 128 * 2);
  float* ACCb  = (float*)alloc((size_t)EP  * 128 * 4);
  float* r1    = (float*)alloc((size_t)EP  * 128 * 4);
  float* r2    = (float*)alloc((size_t)EP  * 128 * 4);
  float* xaA   = (float*)alloc((size_t)NAp * 128 * 4);
  float* xaB   = (float*)alloc((size_t)NAp * 128 * 4);
  h16*   XS    = (h16*)  alloc((size_t)NAp * 128 * 2);
  h16*   XD    = (h16*)  alloc((size_t)NAp * 128 * 2);
  h16*   Kb    = (h16*)  alloc((size_t)NAp * 128 * 2);
  h16*   Vb    = (h16*)  alloc((size_t)NAp * 128 * 2);
  h16*   Qb    = (h16*)  alloc((size_t)NAp * 128 * 2);
  h16*   RN    = (h16*)  alloc((size_t)EP  * 128 * 2);
  h16*   KR    = (h16*)  alloc((size_t)EP  * 128 * 2);
  h16*   VR    = (h16*)  alloc((size_t)EP  * 128 * 2);
  float* simb  = (float*)alloc((size_t)EMX * 8 * 4);
  float* mb    = (float*)alloc((size_t)NA  * 8 * 4);
  float* den   = (float*)alloc((size_t)NA  * 8 * 4);
  float* msg   = (float*)alloc((size_t)NAp * 128 * 4);
  h16*   CAT   = (h16*)  alloc((size_t)NAp * 256 * 2);
  float* Sb    = (float*)alloc((size_t)NAp * 128 * 4);
  float* Gb    = (float*)alloc((size_t)NAp * 128 * 4);
  h16*   AGG   = (h16*)  alloc((size_t)NAp * 128 * 2);
  float* AO    = (float*)alloc((size_t)NAp * 128 * 4);
  float* Xt    = (float*)alloc((size_t)NAp * 128 * 4);
  h16*   HN    = (h16*)  alloc((size_t)NAp * 128 * 2);
  h16*   FH    = (h16*)  alloc((size_t)NAp * 512 * 2);
  float* FO    = (float*)alloc((size_t)NAp * 128 * 4);

  auto gemm = [&](const h16* A, int lda, const h16* Bt, int Mp, int K, int N,
                  const float* bias, void* Cout, float* Cacc,
                  const float* ls, const float* lb, int epi) {
    dim3 g((unsigned)(Mp / 64), (unsigned)(N / 128));
    k_gemm<<<g, 128, 0, stream>>>(A, lda, Bt, K, N, bias, Cout, Cacc, ls, lb, epi);
  };
  auto fill = [&](float* p, long n, float v) {
    k_fill<<<cdiv(n, 256), 256, 0, stream>>>(p, n, v);
  };

  // ---- raw features ----
  k_agent_feat<<<cdiv(NA, 256), 256, 0, stream>>>(agents, xcont, NA);
  k_edge_r1<<<cdiv(E1, 256), 256, 0, stream>>>(pos_pl, orient_pl, agents, es1, ed1, r1c, E1);
  k_edge_r2<<<cdiv(E2, 256), 256, 0, stream>>>(agents, es2, ed2, r2c, E2);

  // ---- Fourier embedding MLPs (all WMMA) ----
  auto run_four = [&](const FourW& f, const float* cont, int M,
                      const int* type, const float* temb, float* OUT) {
    int Mp = pad64(M);
    fill(ACCb, (long)Mp * 128, 0.f);
    for (int d = 0; d < f.dim; ++d) {
      k_fourier_f<<<cdiv((long)M * 64, 256), 256, 0, stream>>>(cont, f.dim, d, f.freqs, Fbuf, M);
      gemm(Fbuf, FPAD, f.W1t + (size_t)d * HDIM * FPAD, Mp, FPAD, 128,
           f.b1 + d * 128, Hbuf, nullptr, f.ln1_s + d * 128, f.ln1_b + d * 128, EPI_LNRELU_F16);
      gemm(Hbuf, 128, f.W2t + (size_t)d * HDIM * HDIM, Mp, 128, 128,
           f.b2 + d * 128, nullptr, ACCb, nullptr, nullptr, EPI_ACC_F32);
    }
    k_postln<<<cdiv(M, 8), 256, 0, stream>>>(ACCb, type, temb, f.oln_s, f.oln_b, Hbuf, M);
    gemm(Hbuf, 128, f.Outt, Mp, 128, 128, f.out_b, OUT, nullptr, nullptr, nullptr, EPI_F32);
  };
  run_four(fx,  xcont, NA, atype, type_emb, xaA);
  run_four(fr1, r1c,   E1, nullptr, nullptr, r1);
  run_four(fr2, r2c,   E2, nullptr, nullptr, r2);

  // ---- graph attention layer ----
  auto run_attn = [&](const AttnW& w, const float* xsrc, int Ns,
                      const float* xdin, float* xdout, const float* rbuf, int Ecnt,
                      const int* es, const int* ed, bool bip) {
    int NsP = pad64(Ns), NdP = NAp, Ep2 = pad64(Ecnt);
    k_ln<<<cdiv(Ns, 8), 256, 0, stream>>>(xsrc, w.lns_s, w.lns_b, XS, Ns);
    k_ln<<<cdiv(NA, 8), 256, 0, stream>>>(xdin, bip ? w.lnd_s : w.lns_s,
                                          bip ? w.lnd_b : w.lns_b, XD, NA);
    k_ln<<<cdiv(Ecnt, 8), 256, 0, stream>>>(rbuf, w.lnr_s, w.lnr_b, RN, Ecnt);

    gemm(XS, 128, w.Tk,  NsP, 128, 128, nullptr, Kb, nullptr, nullptr, nullptr, EPI_F16);
    gemm(XS, 128, w.Tv,  NsP, 128, 128, nullptr, Vb, nullptr, nullptr, nullptr, EPI_F16);
    gemm(XD, 128, w.Tq,  NdP, 128, 128, w.tq_b,  Qb, nullptr, nullptr, nullptr, EPI_F16);
    gemm(RN, 128, w.Tkr, Ep2, 128, 128, nullptr, KR, nullptr, nullptr, nullptr, EPI_F16);
    gemm(RN, 128, w.Tvr, Ep2, 128, 128, nullptr, VR, nullptr, nullptr, nullptr, EPI_F16);

    fill(mb,  (long)NA * 8,  -3e38f);
    fill(den, (long)NA * 8,   0.f);
    fill(msg, (long)NA * 128, 0.f);
    k_sim<<<cdiv((long)Ecnt * 8, 256), 256, 0, stream>>>(Qb, Kb, KR, es, ed, simb, mb, Ecnt);
    k_exp<<<cdiv((long)Ecnt * 8, 256), 256, 0, stream>>>(simb, ed, mb, den, Ecnt);
    k_msg<<<cdiv((long)Ecnt * 128, 256), 256, 0, stream>>>(simb, den, Vb, VR, es, ed, msg, Ecnt);

    gemm(XD, 128, w.Ts, NdP, 128, 128, w.ts_b, Sb, nullptr, nullptr, nullptr, EPI_F32);
    k_concat<<<cdiv((long)NA * 128, 256), 256, 0, stream>>>(msg, XD, CAT, NA);
    gemm(CAT, 256, w.Tg, NdP, 256, 128, w.tg_b, Gb, nullptr, nullptr, nullptr, EPI_SIG_F32);
    k_agg<<<cdiv((long)NA * 128, 256), 256, 0, stream>>>(msg, Gb, Sb, AGG, NA);
    gemm(AGG, 128, w.Tout, NdP, 128, 128, w.tout_b, AO, nullptr, nullptr, nullptr, EPI_F32);
    k_add_ln<<<cdiv(NA, 8), 256, 0, stream>>>(xdin, AO, w.ap_s, w.ap_b, Xt, NA);

    k_ln<<<cdiv(NA, 8), 256, 0, stream>>>(Xt, w.ffpr_s, w.ffpr_b, HN, NA);
    gemm(HN, 128, w.Tff1, NdP, 128, 512, w.ff1_b, FH, nullptr, nullptr, nullptr, EPI_RELU_F16);
    gemm(FH, 512, w.Tff2, NdP, 512, 128, w.ff2_b, FO, nullptr, nullptr, nullptr, EPI_F32);
    k_add_ln<<<cdiv(NA, 8), 256, 0, stream>>>(Xt, FO, w.ffpo_s, w.ffpo_b, xdout, NA);
  };

  for (int l = 0; l < 2; ++l) {
    run_attn(pl2a[l], x_pl, NPL, xaA, xaB, r1, E1, es1, ed1, true);
    run_attn(a2a[l],  xaB,  NA,  xaB, xaA, r2, E2, es2, ed2, false);
  }

  k_gather<<<cdiv((long)NEG * 128, 256), 256, 0, stream>>>(xaA, ego, (float*)d_out, NEG);
}
